// YuriiFormerModel_15075335209298
// MI455X (gfx1250) — matrix operand order repaired and checked by reading, not compiled, not run on hardware
//
#include <hip/hip_runtime.h>
#include <math.h>

// ---------------------------------------------------------------------------
// Types / helpers
// ---------------------------------------------------------------------------
typedef __bf16 bf16;
typedef __attribute__((ext_vector_type(16))) __bf16 v16bf;
typedef __attribute__((ext_vector_type(8)))  float  v8f;

union FragU {
    uint4 u4[2];
    bf16  bs[16];
    v16bf v;
};

__device__ __forceinline__ v8f wmma_bf16(v16bf a, v16bf b, v8f c) {
    // D(f32 16x16) = A(bf16 16x32) * B(bf16 32x16) + C
    return __builtin_amdgcn_wmma_f32_16x16x32_bf16(false, a, false, b,
                                                   (short)0, c, false, false);
}

// block = 256 threads = 8 wave32
__device__ __forceinline__ float block_sum(float val, float* red) {
    int tid = threadIdx.x;
    #pragma unroll
    for (int m = 16; m >= 1; m >>= 1) val += __shfl_xor(val, m, 32);
    if ((tid & 31) == 0) red[tid >> 5] = val;
    __syncthreads();
    float t = (tid < 8) ? red[tid] : 0.0f;
    if (tid < 32) {
        #pragma unroll
        for (int m = 4; m >= 1; m >>= 1) t += __shfl_xor(t, m, 32);
        if (tid == 0) red[0] = t;
    }
    __syncthreads();
    float r = red[0];
    __syncthreads();
    return r;
}

// ---------------------------------------------------------------------------
// Embedding: x = tok_emb[idx] + pos_emb, v = tok_v0[idx] + pos_v0
// ---------------------------------------------------------------------------
__global__ __launch_bounds__(256) void embed_kernel(
        const int* __restrict__ idx,
        const float* __restrict__ tok,   const float* __restrict__ pos,
        const float* __restrict__ tokv0, const float* __restrict__ posv0,
        float* __restrict__ x, float* __restrict__ v) {
    int row = blockIdx.x;          // 0..2047 (b*1024+t)
    int t   = row & 1023;
    int tk  = idx[row];
    size_t rb = (size_t)row * 768;
    size_t tb = (size_t)tk  * 768;
    size_t pb = (size_t)t   * 768;
    #pragma unroll
    for (int i = 0; i < 3; ++i) {
        int c = threadIdx.x + i * 256;
        x[rb + c] = tok[tb + c]   + pos[pb + c];
        v[rb + c] = tokv0[tb + c] + posv0[pb + c];
    }
}

// ---------------------------------------------------------------------------
// out_bf16 = LayerNorm(x + sigmoid(gate_raw[layer]) * v) * w
// (v==nullptr -> plain LN of x; used for ln_f)
// ---------------------------------------------------------------------------
__global__ __launch_bounds__(256) void ln_axpy_kernel(
        const float* __restrict__ x, const float* __restrict__ v,
        const float* __restrict__ gate_raw, int layer,
        const float* __restrict__ w, bf16* __restrict__ out) {
    __shared__ float red[8];
    int tid = threadIdx.x;
    size_t rb = (size_t)blockIdx.x * 768;
    float mu = 0.0f;
    if (gate_raw) mu = 1.0f / (1.0f + __expf(-gate_raw[layer]));
    float val[3];
    #pragma unroll
    for (int i = 0; i < 3; ++i) {
        int c = tid + i * 256;
        float xv = x[rb + c];
        if (v) xv += mu * v[rb + c];
        val[i] = xv;
    }
    float mean = block_sum(val[0] + val[1] + val[2], red) * (1.0f / 768.0f);
    float q = 0.0f;
    #pragma unroll
    for (int i = 0; i < 3; ++i) { float d = val[i] - mean; q += d * d; }
    float rstd = rsqrtf(block_sum(q, red) * (1.0f / 768.0f) + 1e-5f);
    #pragma unroll
    for (int i = 0; i < 3; ++i) {
        int c = tid + i * 256;
        out[rb + c] = (bf16)((val[i] - mean) * rstd * w[c]);
    }
}

// ---------------------------------------------------------------------------
// g = LayerNorm(sigmoid(beta)*v + softplus(gamma)*dx) * w ;  v = g ; x += g
// ---------------------------------------------------------------------------
__global__ __launch_bounds__(256) void vupdate_kernel(
        float* __restrict__ x, float* __restrict__ v,
        const float* __restrict__ dx,
        const float* __restrict__ beta_raw, const float* __restrict__ gamma_raw,
        int layer, const float* __restrict__ w) {
    __shared__ float red[8];
    int tid = threadIdx.x;
    size_t rb = (size_t)blockIdx.x * 768;
    float beta  = 1.0f / (1.0f + __expf(-beta_raw[layer]));
    float gamma = log1pf(__expf(gamma_raw[layer]));
    float val[3];
    #pragma unroll
    for (int i = 0; i < 3; ++i) {
        int c = tid + i * 256;
        val[i] = beta * v[rb + c] + gamma * dx[rb + c];
    }
    float mean = block_sum(val[0] + val[1] + val[2], red) * (1.0f / 768.0f);
    float q = 0.0f;
    #pragma unroll
    for (int i = 0; i < 3; ++i) { float d = val[i] - mean; q += d * d; }
    float rstd = rsqrtf(block_sum(q, red) * (1.0f / 768.0f) + 1e-5f);
    #pragma unroll
    for (int i = 0; i < 3; ++i) {
        int c = tid + i * 256;
        float g = (val[i] - mean) * rstd * w[c];
        v[rb + c] = g;
        x[rb + c] += g;
    }
}

// ---------------------------------------------------------------------------
// exact GELU, fp32 -> bf16
// ---------------------------------------------------------------------------
__global__ __launch_bounds__(256) void gelu_kernel(
        const float* __restrict__ in, bf16* __restrict__ out, int n) {
    int i0 = (blockIdx.x * 256 + threadIdx.x) * 4;
    #pragma unroll
    for (int j = 0; j < 4; ++j) {
        int i = i0 + j;
        if (i < n) {
            float xx = in[i];
            out[i] = (bf16)(0.5f * xx * (1.0f + erff(xx * 0.70710678118f)));
        }
    }
}

// ---------------------------------------------------------------------------
// GEMM: out[M,N] = A[M,K](bf16) * W[N,K](fp32, converted to bf16)^T
// 128 threads = 4 waves (2x2), each wave computes 32(M) x 64(N)
// = 2x4 WMMA accumulators -> 8 v_wmma per 32-deep K-step, A frags reused x4.
// Block tile = 64(M) x 128(N).  M,N multiples of 64/128; K multiple of 32.
// ---------------------------------------------------------------------------
__global__ __launch_bounds__(128) void gemm_bf16_wmma(
        const bf16* __restrict__ A, const float* __restrict__ W,
        float* __restrict__ outF, bf16* __restrict__ outB,
        int M, int N, int K) {
    int tid  = threadIdx.x;
    int wid  = tid >> 5, lane = tid & 31;
    int n16  = lane & 15, hi = lane >> 4;
    int waveM = blockIdx.y * 64  + (wid >> 1) * 32;
    int waveN = blockIdx.x * 128 + (wid & 1) * 64;
    (void)M;

    v8f acc[2][4] = {};
    for (int k = 0; k < K; k += 32) {
        // ---- A fragments (16x32 bf16, lane m = lane%16, contiguous k (same order as B):
        //      K chunk [hi*16, +16))
        v16bf af[2];
        #pragma unroll
        for (int i = 0; i < 2; ++i) {
            const bf16* ap = A + (size_t)(waveM + i * 16 + n16) * K + k;
            FragU f;
            f.u4[0] = *(const uint4*)(ap + hi * 16);
            f.u4[1] = *(const uint4*)(ap + hi * 16 + 8);
            af[i] = f.v;
        }
        // ---- B fragments (32x16 bf16, lane n = lane%16 picks the
        //      output column = weight row; lane half picks K 0-15 / 16-31, contiguous like A)
        v16bf bfr[4];
        #pragma unroll
        for (int j = 0; j < 4; ++j) {
            const float* wp = W + (size_t)(waveN + j * 16 + n16) * K + k + hi * 16;
            if (k + 32 < K) __builtin_prefetch(wp + 32, 0, 1);   // global_prefetch_b8
            FragU f;
            #pragma unroll
            for (int q = 0; q < 4; ++q) {
                float4 w4 = ((const float4*)wp)[q];
                f.bs[q * 4 + 0] = (bf16)w4.x;
                f.bs[q * 4 + 1] = (bf16)w4.y;
                f.bs[q * 4 + 2] = (bf16)w4.z;
                f.bs[q * 4 + 3] = (bf16)w4.w;
            }
            bfr[j] = f.v;
        }
        #pragma unroll
        for (int i = 0; i < 2; ++i)
            #pragma unroll
            for (int j = 0; j < 4; ++j)
                acc[i][j] = wmma_bf16(af[i], bfr[j], acc[i][j]);
    }

    // ---- store (C layout: VGPR r -> row r + 8*hi, col = lane%16)
    if (outF) {
        #pragma unroll
        for (int i = 0; i < 2; ++i)
            #pragma unroll
            for (int j = 0; j < 4; ++j)
                #pragma unroll
                for (int r = 0; r < 8; ++r)
                    outF[(size_t)(waveM + i * 16 + r + 8 * hi) * N
                         + waveN + j * 16 + n16] = acc[i][j][r];
    } else {
        #pragma unroll
        for (int i = 0; i < 2; ++i)
            #pragma unroll
            for (int j = 0; j < 4; ++j)
                #pragma unroll
                for (int r = 0; r < 8; ++r)
                    outB[(size_t)(waveM + i * 16 + r + 8 * hi) * N
                         + waveN + j * 16 + n16] = (bf16)acc[i][j][r];
    }
}

// ---------------------------------------------------------------------------
// Causal flash attention over qkv (bf16, [B*T, 3C], q|k|v sections).
// grid = (T/16, B*H/4); block = 128 = 4 waves; wave w handles head bh=by*4+w,
// all waves share the same 16-row q tile -> uniform kv loop, legal barriers.
// D = 64, scale = 1/8.
// ---------------------------------------------------------------------------
__global__ __launch_bounds__(128) void attn_kernel(
        const bf16* __restrict__ qkv, bf16* __restrict__ y) {
    const int TC3 = 2304;
    int tid  = threadIdx.x;
    int wid  = tid >> 5, lane = tid & 31;
    int n16  = lane & 15, hi = lane >> 4;
    int qBase = blockIdx.x * 16;
    int bh = blockIdx.y * 4 + wid;
    int b = bh / 12, h = bh % 12;
    const bf16* base = qkv + (size_t)b * 1024 * TC3;

    __shared__ __align__(16) bf16 Pb[4][16 * 32];   // P tile per wave
    __shared__ __align__(16) bf16 Vb[4][32 * 64];   // V tile per wave

    // ---- Q fragments, kept live over the whole kv loop
    v16bf qf[2];
    {
        const bf16* qp = base + (size_t)(qBase + n16) * TC3 + h * 64;
        #pragma unroll
        for (int c = 0; c < 2; ++c) {
            FragU f;
            f.u4[0] = *(const uint4*)(qp + c * 32 + hi * 16);
            f.u4[1] = *(const uint4*)(qp + c * 32 + hi * 16 + 8);
            qf[c] = f.v;
        }
    }

    float mrow[8], lrow[8];
    #pragma unroll
    for (int r = 0; r < 8; ++r) { mrow[r] = -1e30f; lrow[r] = 0.0f; }
    v8f o[4] = {};
    const float scale = 0.125f;
    int kvEnd = qBase + 16;

    for (int kv = 0; kv < kvEnd; kv += 32) {
        // ---- stage V tile [32 x 64] for this wave's head into LDS
        #pragma unroll
        for (int i = 0; i < 8; ++i) {
            int vrow = i * 4 + (lane >> 3);
            int vcol = (lane & 7) * 8;
            *(uint4*)&Vb[wid][vrow * 64 + vcol] =
                *(const uint4*)(base + (size_t)(kv + vrow) * TC3 + 1536 + h * 64 + vcol);
        }

        // ---- S = Q * K^T : two 16x16 column tiles, K-dim (=D) split in 2
        v8f s[2] = {};
        #pragma unroll
        for (int j = 0; j < 2; ++j) {
            #pragma unroll
            for (int c = 0; c < 2; ++c) {
                const bf16* kp = base + (size_t)(kv + j * 16 + n16) * TC3
                               + 768 + h * 64 + c * 32 + hi * 16;
                FragU f;
                f.u4[0] = ((const uint4*)kp)[0];
                f.u4[1] = ((const uint4*)kp)[1];
                s[j] = wmma_bf16(qf[c], f.v, s[j]);
            }
        }

        // ---- causal mask + online softmax; write P (bf16) to LDS
        #pragma unroll
        for (int r = 0; r < 8; ++r) {
            int qg = qBase + r + 8 * hi;
            float v0 = s[0][r] * scale;
            float v1 = s[1][r] * scale;
            if (kv + n16 > qg)      v0 = -1e30f;
            if (kv + 16 + n16 > qg) v1 = -1e30f;
            float mx = fmaxf(v0, v1);
            #pragma unroll
            for (int m = 1; m < 16; m <<= 1) mx = fmaxf(mx, __shfl_xor(mx, m, 32));
            float mnew = fmaxf(mrow[r], mx);
            float corr = __expf(mrow[r] - mnew);
            float p0 = __expf(v0 - mnew);
            float p1 = __expf(v1 - mnew);
            float rs = p0 + p1;
            #pragma unroll
            for (int m = 1; m < 16; m <<= 1) rs += __shfl_xor(rs, m, 32);
            lrow[r] = lrow[r] * corr + rs;
            mrow[r] = mnew;
            #pragma unroll
            for (int t2 = 0; t2 < 4; ++t2) o[t2][r] *= corr;
            Pb[wid][(r + 8 * hi) * 32 + n16]      = (bf16)p0;
            Pb[wid][(r + 8 * hi) * 32 + 16 + n16] = (bf16)p1;
        }
        __syncthreads();

        // ---- re-read P in A-fragment layout
        FragU pf;
        {
            const bf16* pp = &Pb[wid][n16 * 32];
            pf.u4[0] = *(const uint4*)(pp + hi * 16);
            pf.u4[1] = *(const uint4*)(pp + hi * 16 + 8);
        }
        // ---- O += P @ V (four 16-wide d tiles); V gathered column-major
        #pragma unroll
        for (int t2 = 0; t2 < 4; ++t2) {
            FragU vf;
            #pragma unroll
            for (int jj = 0; jj < 16; ++jj)
                vf.bs[jj] = Vb[wid][(hi * 16 + jj) * 64 + t2 * 16 + n16];
            o[t2] = wmma_bf16(pf.v, vf.v, o[t2]);
        }
        __syncthreads();
    }

    // ---- normalize and store y (bf16, [B*T, C], heads concatenated)
    #pragma unroll
    for (int t2 = 0; t2 < 4; ++t2) {
        #pragma unroll
        for (int r = 0; r < 8; ++r) {
            float val = o[t2][r] / lrow[r];
            size_t row = (size_t)(b * 1024 + qBase + r + 8 * hi);
            y[row * 768 + h * 64 + t2 * 16 + n16] = (bf16)val;
        }
    }
}

// ---------------------------------------------------------------------------
// Host orchestration
// ---------------------------------------------------------------------------
extern "C" void kernel_launch(void* const* d_in, const int* in_sizes, int n_in,
                              void* d_out, int out_size, void* d_ws, size_t ws_size,
                              hipStream_t stream) {
    (void)in_sizes; (void)n_in; (void)out_size; (void)ws_size;
    const int BT = 2048, C = 768, C3 = 2304, FF = 3072, Vv = 50304, L = 6;

    const int*   idx    = (const int*)d_in[0];
    const float* tok    = (const float*)d_in[1];
    const float* pos    = (const float*)d_in[2];
    const float* tokv0  = (const float*)d_in[3];
    const float* posv0  = (const float*)d_in[4];
    const float* lnaw   = (const float*)d_in[5];
    const float* cattw  = (const float*)d_in[6];
    const float* cprojw = (const float*)d_in[7];
    const float* lnmw   = (const float*)d_in[8];
    const float* fc1w   = (const float*)d_in[9];
    const float* fc2w   = (const float*)d_in[10];
    const float* lnvw   = (const float*)d_in[11];
    const float* mu1r   = (const float*)d_in[12];
    const float* b1r    = (const float*)d_in[13];
    const float* g1r    = (const float*)d_in[14];
    const float* mu2r   = (const float*)d_in[15];
    const float* b2r    = (const float*)d_in[16];
    const float* g2r    = (const float*)d_in[17];
    const float* lnfw   = (const float*)d_in[18];

    char* wsb = (char*)d_ws;
    size_t off = 0;
    auto alloc = [&](size_t bytes) -> void* {
        void* p = wsb + off;
        off += (bytes + 1023) & ~(size_t)1023;
        return p;
    };
    float* x   = (float*)alloc((size_t)BT * C  * 4);
    float* v   = (float*)alloc((size_t)BT * C  * 4);
    bf16*  hb  = (bf16*) alloc((size_t)BT * C  * 2);
    bf16*  qkvb= (bf16*) alloc((size_t)BT * C3 * 2);
    bf16*  yb  = (bf16*) alloc((size_t)BT * C  * 2);
    float* dx  = (float*)alloc((size_t)BT * C  * 4);
    float* ff  = (float*)alloc((size_t)BT * FF * 4);
    bf16*  ffb = (bf16*) alloc((size_t)BT * FF * 2);

    embed_kernel<<<BT, 256, 0, stream>>>(idx, tok, pos, tokv0, posv0, x, v);

    for (int l = 0; l < L; ++l) {
        // attention half-step
        ln_axpy_kernel<<<BT, 256, 0, stream>>>(x, v, mu1r, l, lnaw + l * C, hb);
        gemm_bf16_wmma<<<dim3(C3 / 128, BT / 64), 128, 0, stream>>>(
            hb, cattw + (size_t)l * C3 * C, nullptr, qkvb, BT, C3, C);
        attn_kernel<<<dim3(64, 6), 128, 0, stream>>>(qkvb, yb);
        gemm_bf16_wmma<<<dim3(C / 128, BT / 64), 128, 0, stream>>>(
            yb, cprojw + (size_t)l * C * C, dx, nullptr, BT, C, C);
        vupdate_kernel<<<BT, 256, 0, stream>>>(x, v, dx, b1r, g1r, l, lnvw + l * C);
        // MLP half-step
        ln_axpy_kernel<<<BT, 256, 0, stream>>>(x, v, mu2r, l, lnmw + l * C, hb);
        gemm_bf16_wmma<<<dim3(FF / 128, BT / 64), 128, 0, stream>>>(
            hb, fc1w + (size_t)l * FF * C, ff, nullptr, BT, FF, C);
        gelu_kernel<<<(BT * FF) / 1024, 256, 0, stream>>>(ff, ffb, BT * FF);
        gemm_bf16_wmma<<<dim3(C / 128, BT / 64), 128, 0, stream>>>(
            ffb, fc2w + (size_t)l * C * FF, dx, nullptr, BT, C, FF);
        vupdate_kernel<<<BT, 256, 0, stream>>>(x, v, dx, b2r, g2r, l, lnvw + l * C);
    }

    // final LN + weight-tied lm_head -> logits fp32 [2048, 50304]
    ln_axpy_kernel<<<BT, 256, 0, stream>>>(x, nullptr, nullptr, 0, lnfw, hb);
    gemm_bf16_wmma<<<dim3(Vv / 128, BT / 64), 128, 0, stream>>>(
        hb, tok, (float*)d_out, nullptr, BT, Vv, C);
}
